// MoE_23476291240448
// MI455X (gfx1250) — matrix operand order repaired and checked by reading, not compiled, not run on hardware
//
#include <hip/hip_runtime.h>
#include <hip/hip_bf16.h>

// ---------------- problem constants ----------------
#define EMB   1024
#define NEXP  8
#define HID   2048
#define NTOK  4096            // 2 * 2048 tokens
#define MTILE 32              // tokens per expert tile
#define MAXTILES (NTOK / MTILE)   // 128 worst-case tiles per expert

typedef __attribute__((ext_vector_type(16))) __bf16 bfx16;
typedef __attribute__((ext_vector_type(8)))  float  f32x8;

union Frag {
    bfx16 v;
    uint4 u[2];
};

// ---------------------------------------------------------------------------
// 1) f32 -> bf16 convert + transpose: W[z][R][C] f32  ->  WT[z][C][R] bf16
//    Makes every WMMA B-fragment two 16B global reads along K, and shrinks
//    all expert weights to 128MB so they stay resident in the 192MB L2.
// ---------------------------------------------------------------------------
__global__ __launch_bounds__(256) void cvt_transpose_kernel(
    const float* __restrict__ W, __bf16* __restrict__ WT, int R, int C)
{
    __shared__ float tile[32][33];
    const int z = blockIdx.z;
    const float*  Wz  = W  + (size_t)z * R * C;
    __bf16*       WTz = WT + (size_t)z * R * C;
    const int r0 = blockIdx.x * 32, c0 = blockIdx.y * 32;
    #pragma unroll
    for (int ii = 0; ii < 4; ++ii) {
        int i = threadIdx.y + ii * 8;
        tile[i][threadIdx.x] = Wz[(size_t)(r0 + i) * C + (c0 + threadIdx.x)];
    }
    __syncthreads();
    #pragma unroll
    for (int ii = 0; ii < 4; ++ii) {
        int i = threadIdx.y + ii * 8;
        WTz[(size_t)(c0 + i) * R + (r0 + threadIdx.x)] = (__bf16)tile[threadIdx.x][i];
    }
}

// ---------------------------------------------------------------------------
// 2) Router: one wave32 per token. 8 dot products via lane-strided loads +
//    shuffle reduction, softmax over 8, top-2 with first-index tie-break
//    (matches jax.lax.top_k). Probs NOT renormalized (matches reference).
// ---------------------------------------------------------------------------
__global__ __launch_bounds__(256) void router_kernel(
    const float* __restrict__ x, const float* __restrict__ Wr,
    const float* __restrict__ br,
    int* __restrict__ topk_i, float* __restrict__ topk_p)
{
    const int tok  = (int)((blockIdx.x * blockDim.x + threadIdx.x) >> 5);
    const int lane = threadIdx.x & 31;
    if (tok >= NTOK) return;

    const float* xr = x + (size_t)tok * EMB;
    float acc[NEXP];
    #pragma unroll
    for (int e = 0; e < NEXP; ++e) acc[e] = 0.f;

    for (int d = lane; d < EMB; d += 32) {
        float xv = xr[d];
        const float4* w4 = (const float4*)(Wr + (size_t)d * NEXP);
        float4 wa = w4[0], wb = w4[1];
        acc[0] += xv * wa.x; acc[1] += xv * wa.y;
        acc[2] += xv * wa.z; acc[3] += xv * wa.w;
        acc[4] += xv * wb.x; acc[5] += xv * wb.y;
        acc[6] += xv * wb.z; acc[7] += xv * wb.w;
    }
    #pragma unroll
    for (int e = 0; e < NEXP; ++e)
        #pragma unroll
        for (int off = 16; off > 0; off >>= 1)
            acc[e] += __shfl_xor(acc[e], off, 32);

    if (lane == 0) {
        float lg[NEXP];
        float mx = -3.0e38f;
        #pragma unroll
        for (int e = 0; e < NEXP; ++e) { lg[e] = acc[e] + br[e]; mx = fmaxf(mx, lg[e]); }
        float s = 0.f;
        #pragma unroll
        for (int e = 0; e < NEXP; ++e) { lg[e] = __expf(lg[e] - mx); s += lg[e]; }
        float inv = 1.f / s;
        #pragma unroll
        for (int e = 0; e < NEXP; ++e) lg[e] *= inv;

        int i1 = 0; float p1 = lg[0];
        #pragma unroll
        for (int e = 1; e < NEXP; ++e) if (lg[e] > p1) { p1 = lg[e]; i1 = e; }
        int i2 = -1; float p2 = -1.f;
        #pragma unroll
        for (int e = 0; e < NEXP; ++e)
            if (e != i1 && lg[e] > p2) { p2 = lg[e]; i2 = e; }

        topk_i[tok * 2 + 0] = i1;  topk_i[tok * 2 + 1] = i2;
        topk_p[tok * 2 + 0] = p1;  topk_p[tok * 2 + 1] = p2;
    }
}

// ---------------------------------------------------------------------------
// 3) Build per-expert token lists. One wave per expert, tokens scanned in
//    index order with ballot+popc prefix sums -> fully deterministic.
// ---------------------------------------------------------------------------
__global__ __launch_bounds__(256) void build_kernel(
    const int* __restrict__ topk_i, const float* __restrict__ topk_p,
    int* __restrict__ counts, int* __restrict__ tok_list,
    int* __restrict__ slot_list, float* __restrict__ gate_list)
{
    const int e    = threadIdx.x >> 5;   // wave == expert
    const int lane = threadIdx.x & 31;
    if (e >= NEXP) return;

    int cnt = 0;
    for (int tb = 0; tb < NTOK; tb += 32) {
        int t  = tb + lane;
        int i0 = topk_i[t * 2 + 0];
        int i1 = topk_i[t * 2 + 1];
        int hit = (i0 == e) ? 1 : ((i1 == e) ? 2 : 0);   // slot+1
        unsigned m32 = (unsigned)__ballot(hit != 0);
        int pos = cnt + __popc(m32 & ((1u << lane) - 1u));
        if (hit) {
            tok_list [e * NTOK + pos] = t;
            slot_list[e * NTOK + pos] = hit - 1;
            gate_list[e * NTOK + pos] = topk_p[t * 2 + (hit - 1)];
        }
        cnt += __popc(m32);
    }
    if (lane == 0) counts[e] = cnt;
}

// ---------------------------------------------------------------------------
// WMMA fragment loaders (layouts per CDNA5 ISA §7.12.2, bf16 16x16x32)
// ---------------------------------------------------------------------------
__device__ __forceinline__ bfx16 load_a_frag(const __bf16* a_row, int k0, int hi)
{
    // lane<16: K = k0+{0..7,16..23}; lane>=16: K = k0+{8..15,24..31}
    Frag f;
    f.u[0] = *(const uint4*)(a_row + k0 + (hi ? 8 : 0));
    f.u[1] = *(const uint4*)(a_row + k0 + 16 + (hi ? 8 : 0));
    return f.v;
}

__device__ __forceinline__ bfx16 load_b_frag(const __bf16* bt_col, int k0, int hi)
{
    // BT is [N][K]: lane holds column n; lane<16: K = k0+{0..7,16..23}; lane>=16: K = k0+{8..15,24..31}
    Frag f;
    f.u[0] = *(const uint4*)(bt_col + k0 + (hi ? 8 : 0));
    f.u[1] = *(const uint4*)(bt_col + k0 + 16 + (hi ? 8 : 0));
    return f.v;
}

// One wave computes a 32(M) x 64(N) fp32 tile: A (32xK bf16) in LDS,
// B from pre-transposed bf16 weights [N][K] in global (L2-resident).
__device__ __forceinline__ void wave_gemm_32x64(
    const __bf16* __restrict__ Alds, int lda,
    const __bf16* __restrict__ BT, int ldb, int n0, int K,
    f32x8 (&acc)[2][4])
{
    const int lane = threadIdx.x & 31;
    const int hi   = lane >> 4;
    const int ml   = lane & 15;
    const __bf16* arow0 = Alds + (size_t)ml * lda;
    const __bf16* arow1 = Alds + (size_t)(16 + ml) * lda;
    const __bf16* bcol[4];
    #pragma unroll
    for (int nt = 0; nt < 4; ++nt)
        bcol[nt] = BT + (size_t)(n0 + nt * 16 + ml) * ldb;

    for (int k0 = 0; k0 < K; k0 += 32) {
        bfx16 a0 = load_a_frag(arow0, k0, hi);
        bfx16 a1 = load_a_frag(arow1, k0, hi);
        #pragma unroll
        for (int nt = 0; nt < 4; ++nt) {
            bfx16 b = load_b_frag(bcol[nt], k0, hi);
            acc[0][nt] = __builtin_amdgcn_wmma_f32_16x16x32_bf16(
                false, a0, false, b, (short)0, acc[0][nt], false, false);
            acc[1][nt] = __builtin_amdgcn_wmma_f32_16x16x32_bf16(
                false, a1, false, b, (short)0, acc[1][nt], false, false);
        }
    }
}

// ---------------------------------------------------------------------------
// 4) Fused expert MLP. One workgroup = (expert, 32-token tile).
//    LDS: bufA (32x2048 bf16, holds x then h2) + bufH (32x2048 bf16, h1)
//    = 256KB of the 320KB WGP LDS. All three GEMMs stay on-chip.
// ---------------------------------------------------------------------------
#define SMEM_MAIN ((size_t)2 * MTILE * HID * 2 + 512)

__global__ __launch_bounds__(256, 1) void moe_expert_kernel(
    const float* __restrict__ x,
    const float* __restrict__ b1, const float* __restrict__ bg,
    const float* __restrict__ b2,
    const __bf16* __restrict__ W1T, const __bf16* __restrict__ WgT,
    const __bf16* __restrict__ W2T,
    const int* __restrict__ counts, const int* __restrict__ tok_list,
    const int* __restrict__ slot_list, const float* __restrict__ gate_list,
    float* __restrict__ yc)
{
    extern __shared__ char smem[];
    __bf16* bufA   = (__bf16*)smem;                         // 32x2048 bf16
    __bf16* bufH   = (__bf16*)(smem + (size_t)MTILE * HID * 2);
    float*  gatesS = (float*)(smem + (size_t)2 * MTILE * HID * 2);
    int*    tokS   = (int*)(gatesS + MTILE);
    int*    slotS  = (int*)(tokS + MTILE);

    const int e    = blockIdx.x / MAXTILES;
    const int tile = blockIdx.x % MAXTILES;
    const int cnt  = counts[e];
    const int base = tile * MTILE;
    if (base >= cnt) return;

    const int wid  = threadIdx.x >> 5;
    const int lane = threadIdx.x & 31;

    // ---- stage token metadata ----
    if (threadIdx.x < MTILE) {
        int i = base + (int)threadIdx.x;
        bool v = i < cnt;
        tokS [threadIdx.x] = v ? tok_list [e * NTOK + i] : 0;
        slotS[threadIdx.x] = v ? slot_list[e * NTOK + i] : 0;
        gatesS[threadIdx.x] = v ? gate_list[e * NTOK + i] : 0.f;
    }
    __syncthreads();

    // ---- gather x rows -> bufA (bf16, 32 x 1024) ----
    {
        int r = threadIdx.x >> 3;
        int cband = (threadIdx.x & 7) * 128;
        const float* xr = x + (size_t)tokS[r] * EMB + cband;
        __bf16* dst = bufA + (size_t)r * EMB + cband;
        #pragma unroll 8
        for (int j = 0; j < 128; j += 4) {
            float4 f = *(const float4*)(xr + j);
            dst[j + 0] = (__bf16)f.x; dst[j + 1] = (__bf16)f.y;
            dst[j + 2] = (__bf16)f.z; dst[j + 3] = (__bf16)f.w;
        }
    }
    __syncthreads();

    f32x8 zero = {0.f, 0.f, 0.f, 0.f, 0.f, 0.f, 0.f, 0.f};
    const int hi = lane >> 4, nc = lane & 15;

    // ---- GEMM1: h1 = x @ W1 + b1  (K=1024, N=2048) ----
    {
        const __bf16* W1e = W1T + (size_t)e * HID * EMB;
        const float*  b1e = b1 + (size_t)e * HID;
        for (int ss = 0; ss < 4; ++ss) {
            int n0 = wid * 256 + ss * 64;
            f32x8 acc[2][4];
            #pragma unroll
            for (int i = 0; i < 2; ++i)
                #pragma unroll
                for (int j = 0; j < 4; ++j) acc[i][j] = zero;
            wave_gemm_32x64(bufA, EMB, W1e, EMB, n0, EMB, acc);
            #pragma unroll
            for (int mt = 0; mt < 2; ++mt)
                #pragma unroll
                for (int nt = 0; nt < 4; ++nt) {
                    int n = n0 + nt * 16 + nc;
                    float bias = b1e[n];
                    #pragma unroll
                    for (int r = 0; r < 8; ++r) {
                        int m = mt * 16 + hi * 8 + r;
                        bufH[(size_t)m * HID + n] = (__bf16)(acc[mt][nt][r] + bias);
                    }
                }
        }
    }
    __syncthreads();

    // ---- GEMM2: h2 = relu(h1 @ Wg + bg)  (K=2048, N=2048) ----
    {
        const __bf16* Wge = WgT + (size_t)e * HID * HID;
        const float*  bge = bg + (size_t)e * HID;
        for (int ss = 0; ss < 4; ++ss) {
            int n0 = wid * 256 + ss * 64;
            f32x8 acc[2][4];
            #pragma unroll
            for (int i = 0; i < 2; ++i)
                #pragma unroll
                for (int j = 0; j < 4; ++j) acc[i][j] = zero;
            wave_gemm_32x64(bufH, HID, Wge, HID, n0, HID, acc);
            #pragma unroll
            for (int mt = 0; mt < 2; ++mt)
                #pragma unroll
                for (int nt = 0; nt < 4; ++nt) {
                    int n = n0 + nt * 16 + nc;
                    float bias = bge[n];
                    #pragma unroll
                    for (int r = 0; r < 8; ++r) {
                        int m = mt * 16 + hi * 8 + r;
                        float v = acc[mt][nt][r] + bias;
                        bufA[(size_t)m * HID + n] = (__bf16)(v > 0.f ? v : 0.f);
                    }
                }
        }
    }
    __syncthreads();

    // ---- GEMM3: y = (h2 @ W2 + b2) * gate -> yc[slot][token]  (K=2048, N=1024) ----
    {
        const __bf16* W2e = W2T + (size_t)e * EMB * HID;
        const float*  b2e = b2 + (size_t)e * EMB;
        for (int ss = 0; ss < 2; ++ss) {
            int n0 = wid * 128 + ss * 64;
            f32x8 acc[2][4];
            #pragma unroll
            for (int i = 0; i < 2; ++i)
                #pragma unroll
                for (int j = 0; j < 4; ++j) acc[i][j] = zero;
            wave_gemm_32x64(bufA, HID, W2e, HID, n0, HID, acc);
            #pragma unroll
            for (int mt = 0; mt < 2; ++mt)
                #pragma unroll
                for (int nt = 0; nt < 4; ++nt) {
                    int n = n0 + nt * 16 + nc;
                    float bias = b2e[n];
                    #pragma unroll
                    for (int r = 0; r < 8; ++r) {
                        int m = mt * 16 + hi * 8 + r;
                        if (base + m < cnt) {
                            float v = (acc[mt][nt][r] + bias) * gatesS[m];
                            yc[((size_t)slotS[m] * NTOK + tokS[m]) * EMB + n] = v;
                        }
                    }
                }
        }
    }
}

// ---------------------------------------------------------------------------
// 5) Deterministic combine: every token has exactly 2 (distinct) experts,
//    so both slots are always written. Fixed add order -> bitwise stable.
// ---------------------------------------------------------------------------
__global__ __launch_bounds__(256) void combine_kernel(
    const float* __restrict__ yc, float* __restrict__ out)
{
    size_t i = (size_t)blockIdx.x * blockDim.x + threadIdx.x;
    out[i] = yc[i] + yc[(size_t)NTOK * EMB + i];
}

// ---------------------------------------------------------------------------
extern "C" void kernel_launch(void* const* d_in, const int* in_sizes, int n_in,
                              void* d_out, int out_size, void* d_ws, size_t ws_size,
                              hipStream_t stream)
{
    (void)in_sizes; (void)n_in; (void)out_size; (void)ws_size;
    const float* x  = (const float*)d_in[0];
    const float* Wr = (const float*)d_in[1];
    const float* br = (const float*)d_in[2];
    const float* W1 = (const float*)d_in[3];
    const float* b1 = (const float*)d_in[4];
    const float* Wg = (const float*)d_in[5];
    const float* bg = (const float*)d_in[6];
    const float* W2 = (const float*)d_in[7];
    const float* b2 = (const float*)d_in[8];
    float* out = (float*)d_out;

    char* ws = (char*)d_ws;
    size_t off = 0;
    auto alloc = [&](size_t bytes) -> void* {
        off = (off + 255) & ~(size_t)255;
        void* p = ws + off;
        off += bytes;
        return p;
    };

    __bf16* W1T = (__bf16*)alloc((size_t)NEXP * EMB * HID * 2);
    __bf16* WgT = (__bf16*)alloc((size_t)NEXP * HID * HID * 2);
    __bf16* W2T = (__bf16*)alloc((size_t)NEXP * HID * EMB * 2);
    int*    topk_i    = (int*)  alloc((size_t)NTOK * 2 * 4);
    float*  topk_p    = (float*)alloc((size_t)NTOK * 2 * 4);
    int*    counts    = (int*)  alloc((size_t)NEXP * 4);
    int*    tok_list  = (int*)  alloc((size_t)NEXP * NTOK * 4);
    int*    slot_list = (int*)  alloc((size_t)NEXP * NTOK * 4);
    float*  gate_list = (float*)alloc((size_t)NEXP * NTOK * 4);
    float*  yc        = (float*)alloc((size_t)2 * NTOK * EMB * 4);

    dim3 blkT(32, 8, 1);
    // W1: [E][1024][2048] -> [E][2048][1024]
    cvt_transpose_kernel<<<dim3(EMB / 32, HID / 32, NEXP), blkT, 0, stream>>>(W1, W1T, EMB, HID);
    // Wg: [E][2048][2048]
    cvt_transpose_kernel<<<dim3(HID / 32, HID / 32, NEXP), blkT, 0, stream>>>(Wg, WgT, HID, HID);
    // W2: [E][2048][1024] -> [E][1024][2048]
    cvt_transpose_kernel<<<dim3(HID / 32, EMB / 32, NEXP), blkT, 0, stream>>>(W2, W2T, HID, EMB);

    router_kernel<<<dim3((NTOK * 32) / 256), dim3(256), 0, stream>>>(x, Wr, br, topk_i, topk_p);

    build_kernel<<<dim3(1), dim3(256), 0, stream>>>(topk_i, topk_p, counts,
                                                    tok_list, slot_list, gate_list);

    moe_expert_kernel<<<dim3(NEXP * MAXTILES), dim3(256), SMEM_MAIN, stream>>>(
        x, b1, bg, b2, W1T, WgT, W2T,
        counts, tok_list, slot_list, gate_list, yc);

    combine_kernel<<<dim3((NTOK * EMB) / 256), dim3(256), 0, stream>>>(yc, out);
}
